// EncoderLayer_17274358465005
// MI455X (gfx1250) — matrix operand order repaired and checked by reading, not compiled, not run on hardware
//
#include <hip/hip_runtime.h>
#include <stdint.h>

typedef __attribute__((ext_vector_type(16))) __bf16 bf16x16;
typedef __attribute__((ext_vector_type(8)))  float  floatx8;

union FragU {
  struct { uint4 lo, hi; } u;
  bf16x16 v;
};

__device__ __forceinline__ unsigned pack2bf(float a, float b) {
  unsigned ua = __float_as_uint(a);
  unsigned ub = __float_as_uint(b);
  ua = (ua + 0x7FFFu + ((ua >> 16) & 1u)) >> 16;   // round-to-nearest-even
  ub = (ub + 0x7FFFu + ((ub >> 16) & 1u)) >> 16;
  return (ua & 0xFFFFu) | (ub << 16);
}

__device__ __forceinline__ unsigned short f2bf(float a) {
  unsigned ua = __float_as_uint(a);
  return (unsigned short)((ua + 0x7FFFu + ((ua >> 16) & 1u)) >> 16);
}

#if defined(__has_builtin)
#if __has_builtin(__builtin_amdgcn_tensor_load_to_lds)
#define HAVE_TDM 1
#endif
#endif

#ifdef HAVE_TDM
// 2D TDM copy: 128 rows x 64 bf16 (128B), row stride 512 elems -> LDS tile.
// D# per CDNA5 ISA ch.8: group0 {count=1, lds_addr, global_addr, type=2},
// group1 {data_size=1(2B), tensor_dim0=64, tensor_dim1=128, tile 64x128,
// dim0_stride=512}. Groups 2/3 zero (2D).
__device__ __forceinline__ void tdm_load_2d(unsigned ldsOff, const void* gptr) {
  typedef __attribute__((ext_vector_type(4))) unsigned int u32x4_t;
  typedef __attribute__((ext_vector_type(8))) int i32x8_t;
  typedef __attribute__((ext_vector_type(4))) int i32x4_t;
  unsigned long long ga = (unsigned long long)(uintptr_t)gptr;
  u32x4_t g0 = { 1u, ldsOff, (unsigned)(ga & 0xFFFFFFFFull),
                 (unsigned)((ga >> 32) & 0x01FFFFFFull) | (2u << 30) };
  i32x8_t g1 = { (int)(1u << 16),    // data_size=1 (2 bytes) at [17:16]
                 (int)(64u << 16),   // tensor_dim0=64 (bits 48..63)
                 (int)(128u << 16),  // tensor_dim1=128 (bits 80..95)
                 (int)(64u << 16),   // tile_dim0=64 (bits 112..127)
                 128,                // tile_dim1=128 (bits 128..143)
                 512,                // tensor_dim0_stride=512 (bits 160..191)
                 0, 0 };
  i32x4_t z4 = { 0, 0, 0, 0 };
#if __clang_major__ >= 23
  i32x8_t z8 = { 0, 0, 0, 0, 0, 0, 0, 0 };
  __builtin_amdgcn_tensor_load_to_lds(g0, g1, z4, z4, z8, 0);
#else
  __builtin_amdgcn_tensor_load_to_lds(g0, g1, z4, z4, 0);
#endif
}
#endif

// ---------------------------------------------------------------------------
// f32 -> bf16 bulk convert (8 elems / thread, b128 in/out)
// ---------------------------------------------------------------------------
__global__ __launch_bounds__(256) void conv_f32_bf16(
    const float* __restrict__ X, unsigned short* __restrict__ Y, int n)
{
  int i = (blockIdx.x * 256 + threadIdx.x) * 8;
  if (i >= n) return;
  float4 a = *(const float4*)(X + i);
  float4 b = *(const float4*)(X + i + 4);
  uint4 o = make_uint4(pack2bf(a.x, a.y), pack2bf(a.z, a.w),
                       pack2bf(b.x, b.y), pack2bf(b.z, b.w));
  *(uint4*)(Y + i) = o;
}

// ---------------------------------------------------------------------------
// One-time weight transform: W f32 [K][N] -> Wt bf16 [N][K] (fragment-ready).
// ---------------------------------------------------------------------------
__global__ __launch_bounds__(256) void w_transpose_bf16(
    const float* __restrict__ W, unsigned short* __restrict__ Wt, int K, int N)
{
  __shared__ __align__(16) unsigned short T[32][32];
  const int tid   = threadIdx.x;
  const int kBase = blockIdx.y * 32;
  const int nBase = blockIdx.x * 32;
  {
    int kl = tid >> 3;
    int n4 = (tid & 7) * 4;
    float4 f = *(const float4*)(W + (size_t)(kBase + kl) * N + nBase + n4);
    T[n4 + 0][kl] = f2bf(f.x);
    T[n4 + 1][kl] = f2bf(f.y);
    T[n4 + 2][kl] = f2bf(f.z);
    T[n4 + 3][kl] = f2bf(f.w);
  }
  __syncthreads();
  {
    int nl = tid >> 3;
    int kq = (tid & 7) * 4;
    uint2 v = *(const uint2*)&T[nl][kq];
    *(uint2*)(Wt + (size_t)(nBase + nl) * K + kBase + kq) = v;
  }
}

// ---------------------------------------------------------------------------
// GEMM: C[M,N] = op(A_bf16[M,K] @ Wt_bf16[N][K] + bias). No LDS, no barriers:
// A fragments per-wave from global (no cross-wave reuse in 128x64 tile),
// B fragments from L2-resident transformed weights, prefetched ahead.
// Each wave: 16x64 strip -> 1 A-frag x 4 B-frags -> 4 WMMAs per K-chunk.
// ---------------------------------------------------------------------------
template <int K, int N, bool OBF16>
__global__ __launch_bounds__(256) void gemm_bf16_wmma(
    const unsigned short* __restrict__ A, const unsigned short* __restrict__ Wt,
    const float* __restrict__ bias, void* __restrict__ Cv, int M, int relu)
{
  const int tid    = threadIdx.x;
  const int lane   = tid & 31;
  const int wid    = tid >> 5;
  const int l15    = lane & 15;
  const int hiHalf = lane >> 4;
  const int mBase  = blockIdx.y * 128 + wid * 16;
  const int nBase  = blockIdx.x * 64;

  floatx8 acc[4] = {};

  int arow = mBase + l15;
  if (arow >= M) arow = M - 1;            // clamp; stores are guarded
  const int kbA = hiHalf ? 8 : 0;         // A: lanes16-31 hold K=8..15,24..31
  const int kbB = hiHalf ? 8 : 0;         // B: lanes16-31 hold K=8..15,24..31
  const unsigned short* ap = A + (size_t)arow * K + kbA;
  const unsigned short* bp[4];
  #pragma unroll
  for (int nt = 0; nt < 4; ++nt)
    bp[nt] = Wt + (size_t)(nBase + nt * 16 + l15) * K + kbB;

  #pragma unroll 4
  for (int kt = 0; kt < K; kt += 32) {
    FragU af;
    af.u.lo = *(const uint4*)(ap);
    af.u.hi = *(const uint4*)(ap + 16);
    ap += 32;
    #pragma unroll
    for (int nt = 0; nt < 4; ++nt) {
      __builtin_prefetch(bp[nt] + 64, 0, 0);   // global_prefetch, 2 chunks ahead
      FragU bf_;
      bf_.u.lo = *(const uint4*)(bp[nt]);
      bf_.u.hi = *(const uint4*)(bp[nt] + 16);
      bp[nt] += 32;
      acc[nt] = __builtin_amdgcn_wmma_f32_16x16x32_bf16(
          false, af.v, false, bf_.v, (short)0, acc[nt], false, false);
    }
  }

  #pragma unroll
  for (int nt = 0; nt < 4; ++nt) {
    const int col = nBase + nt * 16 + l15;
    float bv = bias ? bias[col] : 0.f;
    #pragma unroll
    for (int r = 0; r < 8; ++r) {
      int row = mBase + r + hiHalf * 8;
      if (row < M) {
        float v = acc[nt][r] + bv;
        if (relu) v = fmaxf(v, 0.f);
        if (OBF16) ((unsigned short*)Cv)[(size_t)row * N + col] = f2bf(v);
        else       ((float*)Cv)[(size_t)row * N + col] = v;
      }
    }
  }
}

// ---------------------------------------------------------------------------
// Block-sparse attention, one block per (b, var, h); all-bf16 q/k/v inputs.
// Keys: 128 own-var patches (TDM-staged) + 7 globals (135, padded to 160).
// ---------------------------------------------------------------------------
#define NVAR   7
#define TPV    129
#define PATCH  128
#define NKEY   135
#define KPAD   160
#define DH     64
#define LTOT   903
#define DMODEL 512
#define HEADS  8

// LDS carve (single object => offset 0): Kl | Vrow | Vt | Pr
#define KL_OFF   0u          // [160][64] bf16, 20KB
#define VROW_OFF 20480u      // [160][64] bf16, 20KB

__global__ __launch_bounds__(288) void attn_wmma(
    const unsigned short* __restrict__ Q, const unsigned short* __restrict__ Kb,
    const unsigned short* __restrict__ Vb, unsigned short* __restrict__ O)
{
  __shared__ __align__(16) unsigned short SMEM[(KPAD * DH) * 2 + DH * KPAD + 9 * 16 * KPAD];
  unsigned short* Kl   = SMEM;                       // [key][dh]
  unsigned short* Vrow = SMEM + KPAD * DH;           // [key][dh]
  unsigned short* Vt   = SMEM + 2 * KPAD * DH;       // [dh][key]
  unsigned short* Pr   = SMEM + 2 * KPAD * DH + DH * KPAD;  // [wave*16+m][key]

  const int tid    = threadIdx.x;
  const int lane   = tid & 31;
  const int wave   = tid >> 5;     // 0..8 (query tile)
  const int l15    = lane & 15;
  const int hiHalf = lane >> 4;

  int blk = blockIdx.x;
  const int h   = blk % HEADS; blk /= HEADS;
  const int var = blk % NVAR;  blk /= NVAR;
  const int b   = blk;

  const size_t headOff = (size_t)h * DH;
  const size_t varTok  = (size_t)(b * LTOT + var * TPV);

#ifdef HAVE_TDM
  if (wave == 0) {
    tdm_load_2d(KL_OFF,   Kb + varTok * DMODEL + headOff);  // 128 patch K rows
    tdm_load_2d(VROW_OFF, Vb + varTok * DMODEL + headOff);  // 128 patch V rows
  }
#else
  // cooperative fallback: patch rows, 8 x b128 per row
  for (int idx = tid; idx < PATCH * 8; idx += 288) {
    int j = idx >> 3, q4 = (idx & 7) * 8;
    uint4 v = *(const uint4*)(Kb + (varTok + j) * DMODEL + headOff + q4);
    *(uint4*)&Kl[j * DH + q4] = v;
    v = *(const uint4*)(Vb + (varTok + j) * DMODEL + headOff + q4);
    *(uint4*)&Vrow[j * DH + q4] = v;
  }
#endif
  // global tokens (7 rows) for K and Vrow
  for (int idx = tid; idx < NVAR * 8; idx += 288) {
    int g = idx >> 3, q4 = (idx & 7) * 8;
    size_t t = (size_t)(b * LTOT + g * TPV + PATCH);
    uint4 v = *(const uint4*)(Kb + t * DMODEL + headOff + q4);
    *(uint4*)&Kl[(PATCH + g) * DH + q4] = v;
    v = *(const uint4*)(Vb + t * DMODEL + headOff + q4);
    *(uint4*)&Vrow[(PATCH + g) * DH + q4] = v;
  }
#ifdef HAVE_TDM
  if (wave == 0) __builtin_amdgcn_s_wait_tensorcnt(0);
#endif
  __syncthreads();

  // ---- Q fragments direct from global bf16 ----
  FragU aq[2];
  {
    int q = wave * 16 + l15;
    bool valid = q < TPV;
    size_t t = varTok + (valid ? q : 0);
    const unsigned short* qr = Q + t * DMODEL + headOff;
    int kb = hiHalf ? 8 : 0;
    #pragma unroll
    for (int c = 0; c < 2; ++c) {
      aq[c].u.lo = *(const uint4*)(qr + c * 32 + kb);
      aq[c].u.hi = *(const uint4*)(qr + c * 32 + kb + 16);
    }
  }

  // ---- S = Q K^T : 10 key-tiles of 16 ----
  floatx8 sc[10];
  #pragma unroll
  for (int t = 0; t < 10; ++t) {
    floatx8 acc = {};
    const unsigned short* kr = Kl + (t * 16 + l15) * DH;
    int kb = hiHalf ? 8 : 0;
    #pragma unroll
    for (int c = 0; c < 2; ++c) {
      FragU bk;
      bk.u.lo = *(const uint4*)(kr + c * 32 + kb);
      bk.u.hi = *(const uint4*)(kr + c * 32 + kb + 16);
      acc = __builtin_amdgcn_wmma_f32_16x16x32_bf16(false, aq[c].v, false, bk.v,
                                                    (short)0, acc, false, false);
    }
    sc[t] = acc;
  }

  // ---- transpose Vrow -> Vt (zero pad keys >= NKEY) ----
  for (int idx = tid; idx < DH * (KPAD / 2); idx += 288) {
    int d  = idx / (KPAD / 2);
    int jp = idx % (KPAD / 2);
    int j0 = 2 * jp, j1 = 2 * jp + 1;
    unsigned a = (j0 < NKEY) ? Vrow[j0 * DH + d] : 0u;
    unsigned c = (j1 < NKEY) ? Vrow[j1 * DH + d] : 0u;
    *(unsigned*)&Vt[d * KPAD + j0] = a | (c << 16);
  }
  __syncthreads();

  // ---- scale + block-sparse mask ----
  const float scale = 0.125f;   // 1/sqrt(64)
  #pragma unroll
  for (int t = 0; t < 10; ++t) {
    int j = t * 16 + l15;
    #pragma unroll
    for (int r = 0; r < 8; ++r) {
      int q = wave * 16 + r + hiHalf * 8;
      float s = sc[t][r] * scale;
      bool blocked;
      if (j >= NKEY)      blocked = true;                 // pad keys
      else if (q >= TPV)  blocked = false;                // pad rows: discarded
      else if (q < PATCH) blocked = (j >= PATCH);         // patch q: own patches only
      else                blocked = (j >= PATCH) && ((j - PATCH) == var); // global q
      sc[t][r] = blocked ? -1e9f : s;
    }
  }

  // ---- register softmax (row = across 16 lanes of one half) ----
  float mx[8], sm[8];
  #pragma unroll
  for (int r = 0; r < 8; ++r) {
    float m = -3.4e38f;
    #pragma unroll
    for (int t = 0; t < 10; ++t) m = fmaxf(m, sc[t][r]);
    #pragma unroll
    for (int sh = 1; sh < 16; sh <<= 1) m = fmaxf(m, __shfl_xor(m, sh, 32));
    mx[r] = m;
  }
  #pragma unroll
  for (int r = 0; r < 8; ++r) {
    float s = 0.f;
    #pragma unroll
    for (int t = 0; t < 10; ++t) {
      float e = __expf(sc[t][r] - mx[r]);
      sc[t][r] = e;
      s += e;
    }
    #pragma unroll
    for (int sh = 1; sh < 16; sh <<= 1) s += __shfl_xor(s, sh, 32);
    sm[r] = 1.f / s;
  }

  // ---- spill probs (bf16) to per-wave LDS (same-wave: DS in-order) ----
  #pragma unroll
  for (int t = 0; t < 10; ++t) {
    int j = t * 16 + l15;
    #pragma unroll
    for (int r = 0; r < 8; ++r) {
      int m = r + hiHalf * 8;
      Pr[(wave * 16 + m) * KPAD + j] = f2bf(sc[t][r] * sm[r]);
    }
  }

  // ---- O = P V : 4 dh-tiles x 5 key-chunks, bf16 output ----
  #pragma unroll
  for (int nt = 0; nt < 4; ++nt) {
    floatx8 acc = {};
    const unsigned short* vr = Vt + (nt * 16 + l15) * KPAD;
    const unsigned short* pr = Pr + (wave * 16 + l15) * KPAD;
    int kbA = hiHalf ? 8 : 0;
    int kbB = hiHalf ? 8 : 0;
    #pragma unroll
    for (int kc = 0; kc < 5; ++kc) {
      FragU ap, bv;
      ap.u.lo = *(const uint4*)(pr + kc * 32 + kbA);
      ap.u.hi = *(const uint4*)(pr + kc * 32 + kbA + 16);
      bv.u.lo = *(const uint4*)(vr + kc * 32 + kbB);
      bv.u.hi = *(const uint4*)(vr + kc * 32 + kbB + 16);
      acc = __builtin_amdgcn_wmma_f32_16x16x32_bf16(false, ap.v, false, bv.v,
                                                    (short)0, acc, false, false);
    }
    int d = nt * 16 + l15;
    #pragma unroll
    for (int r = 0; r < 8; ++r) {
      int q = wave * 16 + r + hiHalf * 8;
      if (q < TPV) {
        size_t tkn = varTok + q;
        O[tkn * DMODEL + headOff + d] = f2bf(acc[r]);
      }
    }
  }
}

// ---------------------------------------------------------------------------
// Out = LayerNorm(X + R) * W + B ; optional bf16 copy for next GEMM's A.
// ---------------------------------------------------------------------------
__global__ __launch_bounds__(256) void ln_residual(
    const float* __restrict__ X, const float* __restrict__ R,
    const float* __restrict__ W, const float* __restrict__ Bv,
    float* __restrict__ Out, unsigned short* __restrict__ OutBf, int ntok)
{
  int tok = blockIdx.x * 8 + (threadIdx.x >> 5);
  if (tok >= ntok) return;
  int lane = threadIdx.x & 31;
  const float* xr = X + (size_t)tok * 512;
  const float* rr = R + (size_t)tok * 512;
  float v[16];
  float s = 0.f;
  #pragma unroll
  for (int i = 0; i < 16; ++i) {
    int c = lane + 32 * i;
    v[i] = xr[c] + rr[c];
    s += v[i];
  }
  #pragma unroll
  for (int sh = 1; sh < 32; sh <<= 1) s += __shfl_xor(s, sh, 32);
  float mean = s * (1.f / 512.f);
  float vs = 0.f;
  #pragma unroll
  for (int i = 0; i < 16; ++i) { float d = v[i] - mean; vs += d * d; }
  #pragma unroll
  for (int sh = 1; sh < 32; sh <<= 1) vs += __shfl_xor(vs, sh, 32);
  float rstd = rsqrtf(vs * (1.f / 512.f) + 1e-5f);
  #pragma unroll
  for (int i = 0; i < 16; ++i) {
    int c = lane + 32 * i;
    float o = (v[i] - mean) * rstd * W[c] + Bv[c];
    Out[(size_t)tok * 512 + c] = o;
    if (OutBf) OutBf[(size_t)tok * 512 + c] = f2bf(o);
  }
}

// ---------------------------------------------------------------------------
extern "C" void kernel_launch(void* const* d_in, const int* in_sizes, int n_in,
                              void* d_out, int out_size, void* d_ws, size_t ws_size,
                              hipStream_t stream) {
  (void)in_sizes; (void)n_in; (void)out_size; (void)ws_size;
  const float* x    = (const float*)d_in[0];
  const float* Wq   = (const float*)d_in[1];
  const float* bq   = (const float*)d_in[2];
  const float* Wk   = (const float*)d_in[3];
  const float* bk   = (const float*)d_in[4];
  const float* Wv   = (const float*)d_in[5];
  const float* bv   = (const float*)d_in[6];
  const float* Wo   = (const float*)d_in[7];
  const float* bo   = (const float*)d_in[8];
  const float* W1   = (const float*)d_in[9];
  const float* b1   = (const float*)d_in[10];
  const float* W2   = (const float*)d_in[11];
  const float* b2   = (const float*)d_in[12];
  const float* ln1w = (const float*)d_in[13];
  const float* ln1b = (const float*)d_in[14];
  const float* ln2w = (const float*)d_in[15];
  const float* ln2b = (const float*)d_in[16];

  const int B = 16, D = DMODEL, Dff = 2048;
  const int M = B * LTOT;               // 14448 token rows
  const size_t SZ = (size_t)M * D;      // elements per activation buffer

  // f32 residual buffers
  float* attnF = (float*)d_ws;          // attn_out (reused as y2 later)
  float* x1F   = attnF + SZ;
  // bf16 activations
  unsigned short* xbf = (unsigned short*)(x1F + SZ);
  unsigned short* qb  = xbf + SZ;
  unsigned short* kb2 = qb + SZ;
  unsigned short* vb2 = kb2 + SZ;
  unsigned short* hb  = vb2 + SZ;       // M x Dff bf16
  // bf16 weights [N][K]
  unsigned short* wq = hb + (size_t)M * Dff;
  unsigned short* wk = wq + (size_t)D * D;
  unsigned short* wv = wk + (size_t)D * D;
  unsigned short* wo = wv + (size_t)D * D;
  unsigned short* w1 = wo + (size_t)D * D;      // [2048][512]
  unsigned short* w2 = w1 + (size_t)D * Dff;    // [512][2048]
  // reuses (sequential stream keeps these safe and deterministic)
  unsigned short* ob   = xbf;           // x_bf16 dead after QKV
  unsigned short* x1bf = qb;            // q dead after attention
  float*          y2F  = attnF;         // attn_out dead after LN1

  dim3 thr(256);
  dim3 gD(D / 64, (M + 127) / 128);
  dim3 gF(Dff / 64, (M + 127) / 128);
  dim3 gLN((M + 7) / 8);
  dim3 gTsq(D / 32, D / 32);
  dim3 gT1(Dff / 32, D / 32);
  dim3 gT2(D / 32, Dff / 32);

  // one-time transforms
  conv_f32_bf16<<<dim3((M * D) / 2048), thr, 0, stream>>>(x, xbf, M * D);
  w_transpose_bf16<<<gTsq, thr, 0, stream>>>(Wq, wq, D, D);
  w_transpose_bf16<<<gTsq, thr, 0, stream>>>(Wk, wk, D, D);
  w_transpose_bf16<<<gTsq, thr, 0, stream>>>(Wv, wv, D, D);
  w_transpose_bf16<<<gTsq, thr, 0, stream>>>(Wo, wo, D, D);
  w_transpose_bf16<<<gT1,  thr, 0, stream>>>(W1, w1, D, Dff);
  w_transpose_bf16<<<gT2,  thr, 0, stream>>>(W2, w2, Dff, D);

  // QKV projections (bf16 out)
  gemm_bf16_wmma<512, 512, true><<<gD, thr, 0, stream>>>(xbf, wq, bq, qb, M, 0);
  gemm_bf16_wmma<512, 512, true><<<gD, thr, 0, stream>>>(xbf, wk, bk, kb2, M, 0);
  gemm_bf16_wmma<512, 512, true><<<gD, thr, 0, stream>>>(xbf, wv, bv, vb2, M, 0);

  // block-sparse attention (bf16 in/out, TDM-staged K/V)
  attn_wmma<<<dim3(B * NVAR * HEADS), dim3(288), 0, stream>>>(qb, kb2, vb2, ob);

  // output projection (f32 out), residual + LN1 (f32 + bf16 copies)
  gemm_bf16_wmma<512, 512, false><<<gD, thr, 0, stream>>>(ob, wo, bo, attnF, M, 0);
  ln_residual<<<gLN, thr, 0, stream>>>(x, attnF, ln1w, ln1b, x1F, x1bf, M);

  // FFN: relu GEMM (bf16 hidden), then down-proj (f32), residual + LN2 -> out
  gemm_bf16_wmma<512, 2048, true><<<gF, thr, 0, stream>>>(x1bf, w1, b1, hb, M, 1);
  gemm_bf16_wmma<2048, 512, false><<<gD, thr, 0, stream>>>(hb, w2, b2, y2F, M, 0);
  ln_residual<<<gLN, thr, 0, stream>>>(x1F, y2F, ln2w, ln2b, (float*)d_out, nullptr, M);
}
